// CausalSelfAttention_77695958384790
// MI455X (gfx1250) — matrix operand order repaired and checked by reading, not compiled, not run on hardware
//
#include <hip/hip_runtime.h>

typedef _Float16 half16 __attribute__((ext_vector_type(16)));
typedef _Float16 half8  __attribute__((ext_vector_type(8)));
typedef float    float8 __attribute__((ext_vector_type(8)));
// alignment-relaxed view for sliding-window LDS loads (2-byte aligned)
typedef half16 half16_u2 __attribute__((aligned(2)));

#define B_   8
#define T_   256
#define C_   64
#define H_   2
#define HD_  32
#define KW   32
#define PADW 16

static __device__ __forceinline__ float8 wmma_f16(half16 a, half16 b, float8 c) {
  // D = A(16x32 f16) * B(32x16 f16) + C(16x16 f32)
  return __builtin_amdgcn_wmma_f32_16x16x32_f16(false, a, false, b, (short)0, c, false, false);
}

static __device__ __forceinline__ half16 load_a_frag(const _Float16* ap) {
  // two 16B-aligned 8-half chunks: K = kb+0..7 and K = kb+8..15
  half8 lo = *(const half8*)ap;
  half8 hi = *(const half8*)(ap + 8);
  return __builtin_shufflevector(lo, hi, 0,1,2,3,4,5,6,7,8,9,10,11,12,13,14,15);
}

// ---------------- kernel W: w2 -> f16, zero loss accumulator ----------------
__global__ void k_prep(const float* __restrict__ w2, _Float16* __restrict__ w2f,
                       float* __restrict__ loss) {
  int i = blockIdx.x * blockDim.x + threadIdx.x;
  if (i < C_ * HD_ * KW) w2f[i] = (_Float16)w2[i];
  if (i == 0) *loss = 0.f;
}

// ---------------- kernel A: qkv projection + pack q/k/v ---------------------
__global__ void k_qkv(const float* __restrict__ x, const float* __restrict__ W,
                      const float* __restrict__ bias,
                      _Float16* __restrict__ qf, _Float16* __restrict__ kf,
                      _Float16* __restrict__ vf, _Float16* __restrict__ vT,
                      float* __restrict__ v32) {
  __shared__ float xr[C_];
  int bt = blockIdx.x;             // b*T + t
  int b = bt >> 8, t = bt & 255;
  int o = threadIdx.x;             // 0..191
  if (o < C_) xr[o] = x[bt * C_ + o];
  __syncthreads();
  float acc = bias[o];
  for (int i = 0; i < C_; ++i) acc += xr[i] * W[i * (3 * C_) + o];
  if (o < 64) {
    int h = o >> 5, d = o & 31;
    qf[((size_t)(b * H_ + h) * T_ + t) * HD_ + d] = (_Float16)acc;
  } else if (o < 128) {
    int oo = o - 64; int h = oo >> 5, d = oo & 31;
    kf[((size_t)(b * H_ + h) * T_ + t) * HD_ + d] = (_Float16)acc;
  } else {
    int c = o - 128;
    v32[(size_t)bt * C_ + c] = acc;
    int h = c >> 5, d = c & 31;
    vf[((size_t)(b * H_ + h) * T_ + t) * HD_ + d] = (_Float16)acc;
    vT[((size_t)(b * H_ + h) * HD_ + d) * T_ + t] = (_Float16)acc;  // transposed for B-frags
  }
}

// ---------------- kernel B1: S = scale * q.k^T with causal mask (WMMA) ------
__global__ void k_scores(const _Float16* __restrict__ qf, const _Float16* __restrict__ kf,
                         float* __restrict__ att) {
  int bh = blockIdx.x;                              // 16 blocks
  int wave = threadIdx.x >> 5, lane = threadIdx.x & 31;
  const _Float16* q = qf + (size_t)bh * T_ * HD_;
  const _Float16* k = kf + (size_t)bh * T_ * HD_;
  float* S = att + (size_t)bh * T_ * T_;
  const float scale = 0.17677669529663687f;         // 1/sqrt(32)
  int nlo = lane & 15;
  int hi  = (lane & 16) ? 1 : 0;
  int kb  = hi ? 16 : 0;
  for (int i = 0; i < 32; ++i) {
    int tile = wave * 32 + i;                       // 256 tiles / 8 waves
    int Mt = tile >> 4, Nt = tile & 15;
    half16 a = load_a_frag(q + (size_t)(Mt * 16 + nlo) * HD_ + kb);
    // B (32x16): Bmat[d][kk] = k[kk][d]; N=lane&15, K = 16*hi + e (32B-aligned)
    half16 bfr = *(const half16*)(k + (size_t)(Nt * 16 + nlo) * HD_ + hi * 16);
    float8 c = {};
    c = wmma_f16(a, bfr, c);
    int N = Nt * 16 + nlo;
    #pragma unroll
    for (int r = 0; r < 8; ++r) {
      int M = Mt * 16 + hi * 8 + r;
      float v = c[r] * scale;
      if (N > M) v = -3.4028234663852886e38f;       // finfo(f32).min
      S[(size_t)M * T_ + N] = v;
    }
  }
}

// ---------------- kernel B2: row softmax; emit f16 att and f32 att^T --------
__global__ void k_softmax(const float* __restrict__ att, _Float16* __restrict__ attf,
                          float* __restrict__ attT) {
  int row = blockIdx.x * blockDim.x + threadIdx.x;  // 4096 rows (bh,q)
  int bh = row >> 8, q = row & 255;
  const float* r = att + (size_t)row * T_;
  float m = -3.4e38f;
  for (int k = 0; k < T_; ++k) m = fmaxf(m, r[k]);
  float s = 0.f;
  for (int k = 0; k < T_; ++k) s += __expf(r[k] - m);
  float inv = 1.f / s;
  float* tcol = attT + (size_t)bh * T_ * T_;
  _Float16* fr = attf + (size_t)row * T_;
  for (int k = 0; k < T_; ++k) {
    float p = __expf(r[k] - m) * inv;
    fr[k] = (_Float16)p;
    tcol[(size_t)k * T_ + q] = p;                   // transposed copy for conv path
  }
}

// ---------------- kernel B3: target = att @ v (WMMA) ------------------------
__global__ void k_attv(const _Float16* __restrict__ attf, const _Float16* __restrict__ vT,
                       float* __restrict__ target) {
  int bh = blockIdx.x; int b = bh >> 1, h = bh & 1;
  int wave = threadIdx.x >> 5, lane = threadIdx.x & 31;
  int nlo = lane & 15, hi = (lane & 16) ? 1 : 0;
  int kb = hi ? 16 : 0;
  const _Float16* A  = attf + (size_t)bh * T_ * T_;
  const _Float16* Vt = vT   + (size_t)bh * HD_ * T_;
  for (int i = 0; i < 4; ++i) {
    int tile = wave * 4 + i;                        // 32 tiles (16 Mt x 2 Nt) / 8 waves
    int Mt = tile >> 1, Nt = tile & 1;
    float8 acc = {};
    #pragma unroll 2
    for (int ks = 0; ks < 8; ++ks) {                // K = 256
      half16 a = load_a_frag(A + (size_t)(Mt * 16 + nlo) * T_ + ks * 32 + kb);
      // Bmat[kidx][d] = v[kidx][d] = vT[d][kidx]: 16 contiguous halves (32B-aligned)
      half16 bfr = *(const half16*)(Vt + (size_t)(Nt * 16 + nlo) * T_ + ks * 32 + hi * 16);
      acc = wmma_f16(a, bfr, acc);
    }
    #pragma unroll
    for (int r = 0; r < 8; ++r) {
      int qq = Mt * 16 + hi * 8 + r;
      int d  = Nt * 16 + nlo;
      target[((size_t)b * T_ + qq) * C_ + h * HD_ + d] = acc[r];
    }
  }
}

// ------- kernel C: fused conv1 -> conv2(WMMA) -> softmax -> loss -> y -------
// One workgroup (256 thr / 8 waves) per row n of att_r; everything in LDS.
// LDS: 116992 B (< 320 KB/WGP on CDNA5) via dynamic shared memory.
__global__ void __launch_bounds__(256) k_conv_fused(
    const float* __restrict__ attT, const float* __restrict__ w1,
    const float* __restrict__ b1, const _Float16* __restrict__ w2f,
    const float* __restrict__ b2, const float* __restrict__ target,
    const float* __restrict__ v32, float* __restrict__ ypre,
    float* __restrict__ loss) {
  extern __shared__ char smem[];
  float*    r0p = (float*)smem;            // 288 (input row 0, zero-padded 16 each side)
  float*    r1p = r0p + 288;               // 288
  float*    w1s = r1p + 288;               // 2048
  float*    b1s = w1s + 2048;              // 64
  float*    b2s = b1s + 64;                // 64
  float*    red = b2s + 64;                // 64
  float*    srd = red + 64;                // 256 (reduction scratch)
  float*    sr2 = srd + 256;               // 256
  float*    h2s = sr2 + 256;               // 64*257 (stride 257: conflict-free rows)
  _Float16* h1p = (_Float16*)(h2s + 64 * 257);   // [2][32][296] padded f16

  int n = blockIdx.x;
  int b = n >> 8, rem = n & 255;
  int h = rem >> 7, k0 = (rem & 127) * 2;
  int bh = b * H_ + h;
  int tid = threadIdx.x;

  // stage att^T rows (the 2 conv input channels), conv1 weights/biases
  r0p[16 + tid] = attT[(size_t)bh * T_ * T_ + (size_t)k0 * T_ + tid];
  r1p[16 + tid] = attT[(size_t)bh * T_ * T_ + (size_t)(k0 + 1) * T_ + tid];
  if (tid < 16) {
    r0p[tid] = 0.f; r1p[tid] = 0.f;
    r0p[272 + tid] = 0.f; r1p[272 + tid] = 0.f;
  }
  for (int i = 0; i < 8; ++i) w1s[tid + i * 256] = w1[tid + i * 256];
  if (tid < 64) { b1s[tid] = b1[tid]; b2s[tid] = b2[tid]; }
  for (int i = 0; i < 74; ++i) h1p[tid + i * 256] = (_Float16)0.f;  // 18944 halves
  __syncthreads();

  // conv1 (VALU, branch-free via padded inputs): h1[c][l], l in [0,257)
  for (int i = 0; i < 65; ++i) {
    int job = tid + i * 256;
    if (job < 64 * 257) {
      int c = job / 257, l = job % 257;
      int g = c >> 5;
      const float* rg = (g ? r1p : r0p) + l;      // rg[j] = padded_input[l + j - 16 + 16]
      const float* wr = w1s + c * 32;
      float acc = b1s[c];
      #pragma unroll
      for (int j = 0; j < KW; ++j) acc += rg[j] * wr[j];
      h1p[(g * 32 + (c & 31)) * 296 + 16 + l] = (_Float16)fmaxf(acc, 0.f);
    }
  }
  __syncthreads();

  // conv2 as GEMM (WMMA): per group g, (32 x 1024)x(1024 x 256).
  // K-index (ci,j) factors: ci = kstep, j = in-tile offset -> B frag is a
  // 16-half sliding-window read from padded h1 in LDS.
  // ks-outer / 8-Nt-inner: one A fragment feeds 8 WMMAs (8 accumulators).
  int wave = tid >> 5, lane = tid & 31;
  int nlo = lane & 15, hi = (lane & 16) ? 1 : 0;
  int kb = hi ? 16 : 0;
  int rb = hi ? 16 : 0;
  int Mt = wave >> 1;                       // 0..3
  int g  = Mt >> 1;
  int mbase  = (Mt & 1) * 16;
  int ntbase = (wave & 1) * 8;              // this wave's 8 N-tiles
  const _Float16* arow = w2f + (size_t)(g * 32 + mbase + nlo) * 1024;
  const _Float16* brow = h1p + (g * 32) * 296;
  float8 acc[8] = {};
  #pragma unroll 4
  for (int ks = 0; ks < 32; ++ks) {         // K = 1024
    half16 a = load_a_frag(arow + ks * 32 + kb);
    const _Float16* bbase = brow + ks * 296 + rb;
    #pragma unroll
    for (int t = 0; t < 8; ++t) {
      int l0 = (ntbase + t) * 16 + nlo;
      half16 bv = *(const half16_u2*)(bbase + l0);  // sliding window, 2B-aligned
      acc[t] = wmma_f16(a, bv, acc[t]);
    }
  }
  #pragma unroll
  for (int t = 0; t < 8; ++t) {
    int l0 = (ntbase + t) * 16 + nlo;
    #pragma unroll
    for (int r = 0; r < 8; ++r) {
      int c = g * 32 + mbase + hi * 8 + r;
      h2s[c * 257 + l0] = fmaxf(acc[t][r] + b2s[c], 0.f);  // bias + relu
    }
  }
  __syncthreads();

  // per-channel softmax over l, MSE vs target, y[n,c] = sum_l p * v[b,l,c]
  // 4 threads per channel row, LDS tree reductions.
  int c = tid >> 2, part = tid & 3;
  const float* row = h2s + c * 257 + part * 64;
  float lm = -3.4e38f;
  for (int l = 0; l < 64; ++l) lm = fmaxf(lm, row[l]);
  srd[tid] = lm;
  __syncthreads();
  float m = fmaxf(fmaxf(srd[c * 4 + 0], srd[c * 4 + 1]),
                  fmaxf(srd[c * 4 + 2], srd[c * 4 + 3]));
  __syncthreads();
  float ls = 0.f;
  for (int l = 0; l < 64; ++l) ls += __expf(row[l] - m);
  srd[tid] = ls;
  __syncthreads();
  float s = srd[c * 4 + 0] + srd[c * 4 + 1] + srd[c * 4 + 2] + srd[c * 4 + 3];
  float inv = 1.f / s;
  __syncthreads();
  float tgt = target[(size_t)n * C_ + c];
  const float* vb = v32 + (size_t)b * T_ * C_ + (size_t)(part * 64) * C_ + c;
  float lacc = 0.f, yacc = 0.f;
  for (int l = 0; l < 64; ++l) {
    float p = __expf(row[l] - m) * inv;
    float d = p - tgt;
    lacc += d * d;
    yacc += p * vb[(size_t)l * C_];
  }
  srd[tid] = yacc; sr2[tid] = lacc;
  __syncthreads();
  if (part == 0) {
    ypre[(size_t)n * C_ + c] =
        srd[c * 4] + srd[c * 4 + 1] + srd[c * 4 + 2] + srd[c * 4 + 3];
    red[c] = sr2[c * 4] + sr2[c * 4 + 1] + sr2[c * 4 + 2] + sr2[c * 4 + 3];
  }
  __syncthreads();
  if (tid == 0) {
    float t = 0.f;
    for (int i = 0; i < 64; ++i) t += red[i];
    atomicAdd(loss, t);
  }
}

// ---------------- kernel D: y = ypre @ c_proj_w + b; finalize loss ----------
__global__ void k_proj(const float* __restrict__ ypre, const float* __restrict__ W,
                       const float* __restrict__ bias, const float* __restrict__ loss,
                       float* __restrict__ out) {
  int o = blockIdx.x * blockDim.x + threadIdx.x;  // 131072
  int nn = o >> 6, c = o & 63;
  const float* yr = ypre + (size_t)nn * C_;
  float acc = bias[c];
  for (int i = 0; i < C_; ++i) acc += yr[i] * W[i * C_ + c];
  out[o] = acc;
  if (o == 0) out[B_ * T_ * C_] = (*loss) * (1.f / 33554432.f); // mean over 2048*64*256
}

extern "C" void kernel_launch(void* const* d_in, const int* in_sizes, int n_in,
                              void* d_out, int out_size, void* d_ws, size_t ws_size,
                              hipStream_t stream) {
  (void)in_sizes; (void)n_in; (void)out_size; (void)ws_size;
  const float* x     = (const float*)d_in[0];
  const float* attnW = (const float*)d_in[1];
  const float* attnB = (const float*)d_in[2];
  const float* projW = (const float*)d_in[3];
  const float* projB = (const float*)d_in[4];
  const float* w1    = (const float*)d_in[5];
  const float* b1    = (const float*)d_in[6];
  const float* w2    = (const float*)d_in[7];
  const float* b2    = (const float*)d_in[8];
  float* out = (float*)d_out;

  char* ws = (char*)d_ws;
  size_t off = 0;
  auto take = [&](size_t bytes) {
    char* p = ws + off;
    off = (off + bytes + 255) & ~(size_t)255;
    return p;
  };
  _Float16* qf     = (_Float16*)take((size_t)B_ * H_ * T_ * HD_ * 2);   // 256 KB
  _Float16* kf     = (_Float16*)take((size_t)B_ * H_ * T_ * HD_ * 2);
  _Float16* vf     = (_Float16*)take((size_t)B_ * H_ * T_ * HD_ * 2);
  _Float16* vT     = (_Float16*)take((size_t)B_ * H_ * T_ * HD_ * 2);
  float*    v32    = (float*)   take((size_t)B_ * T_ * C_ * 4);         // 512 KB
  float*    att    = (float*)   take((size_t)B_ * H_ * T_ * T_ * 4);    // 4 MB
  _Float16* attf   = (_Float16*)take((size_t)B_ * H_ * T_ * T_ * 2);    // 2 MB
  float*    attT   = (float*)   take((size_t)B_ * H_ * T_ * T_ * 4);    // 4 MB
  float*    target = (float*)   take((size_t)B_ * T_ * C_ * 4);
  float*    ypre   = (float*)   take((size_t)B_ * T_ * C_ * 4);
  _Float16* w2f    = (_Float16*)take((size_t)C_ * HD_ * KW * 2);        // 128 KB
  float*    lossp  = (float*)   take(256);

  k_prep   <<<256,  256, 0, stream>>>(w2, w2f, lossp);
  k_qkv    <<<B_ * T_, 192, 0, stream>>>(x, attnW, attnB, qf, kf, vf, vT, v32);
  k_scores <<<B_ * H_, 256, 0, stream>>>(qf, kf, att);
  k_softmax<<<B_ * H_, 256, 0, stream>>>(att, attf, attT);
  k_attv   <<<B_ * H_, 256, 0, stream>>>(attf, vT, target);

  size_t smem = (size_t)(288 + 288 + 2048 + 64 + 64 + 64 + 256 + 256 + 64 * 257) * 4
              + (size_t)2 * 32 * 296 * 2;   // 116992 B
  k_conv_fused<<<B_ * T_, 256, smem, stream>>>(attT, w1, b1, w2f, b2, target,
                                               v32, ypre, lossp);
  k_proj   <<<(B_ * T_ * C_) / 256, 256, 0, stream>>>(ypre, projW, projB, lossp, out);
}
